// Speller_1116691497181
// MI455X (gfx1250) — matrix operand order repaired and checked by reading, not compiled, not run on hardware
//
#include <hip/hip_runtime.h>
#include <hip/hip_bf16.h>
#include <math.h>

// ---------------- problem constants ----------------
#define NUM_CLASS 2000
#define HID 512
#define NUM_LAYERS 2
#define NUM_HEAD 8
#define ATTN_DIM 64
#define BB 32
#define SS 128
#define TT 1024

typedef __attribute__((ext_vector_type(16))) __bf16 v16bf;
typedef __attribute__((ext_vector_type(8)))  __bf16 v8bf;
typedef __attribute__((ext_vector_type(8)))  float  v8f;

// ---------------- CDNA5 async global->LDS helpers (ASYNCcnt path) ----------------
__device__ __forceinline__ void async_copy_b128(const __bf16* g, unsigned lds_off) {
    // LDS[lds_off .. +15] = MEM[g .. +15], tracked by ASYNCcnt
    asm volatile("global_load_async_to_lds_b128 %0, %1, off"
                 :: "v"(lds_off), "v"((unsigned long long)g)
                 : "memory");
}
__device__ __forceinline__ void wait_async0() {
    asm volatile("s_wait_asynccnt 0x0" ::: "memory");
}

// ---------------- small elementwise kernels ----------------

__global__ void cast_f32_bf16(const float* __restrict__ src, __bf16* __restrict__ dst, long long n) {
    long long i = (long long)blockIdx.x * blockDim.x + threadIdx.x;
    if (i < n) dst[i] = (__bf16)src[i];
}

// x[b,s,:] = bf16(emb[tokens[b,s],:])
__global__ void gather_emb(const int* __restrict__ tok, const float* __restrict__ emb,
                           __bf16* __restrict__ x, int total) {
    int i = blockIdx.x * blockDim.x + threadIdx.x;
    if (i < total) {
        int row = i >> 9;          // HID == 512
        int k   = i & 511;
        x[i] = (__bf16)emb[(long long)tok[row] * HID + k];
    }
}

// ---------------- generic WMMA GEMM (NT) with async-LDS B staging ----------------
// C[m,n] = sum_k A[m,k]*B[n,k] + bias[n]   A,B bf16 row-major.
// Outputs (any may be null): Cf f32, Cb bf16 [m*ldc+n], Ct bf16 transposed [n*ldct+m].
// flags bit1: tanh epilogue.
// Batched: z = blockIdx.z; z1 = z/batch2, z2 = z%batch2; offsets z1*S1+z2*S2.
// Block = 128 threads (4 waves); wave tile 16(M)x64(N); block tile 64x64.
// All 4 waves share one 64x32 B panel -> staged once per k-step via
// global_load_async_to_lds_b128 with double buffering.
// FULL=true: N % 64 == 0, no guards in the hot loop.
#define FLAG_TANH 2

template <bool FULL>
__global__ __launch_bounds__(128, 4)
void gemm_wmma(const __bf16* __restrict__ A, long long lda, long long aS1, long long aS2,
               const __bf16* __restrict__ Bm, long long ldb, long long bS1, long long bS2,
               const float* __restrict__ bias,
               float* __restrict__ Cf, __bf16* __restrict__ Cb, __bf16* __restrict__ Ct,
               long long ldc, long long cS1, long long cS2,
               long long ldct, long long ctS1, long long ctS2,
               int M, int N, int K, int batch2, int flags)
{
    __shared__ __align__(32) __bf16 bpanel[2][64][32];

    int z  = blockIdx.z;
    int z1 = z / batch2, z2 = z % batch2;
    A  += z1 * aS1 + z2 * aS2;
    Bm += z1 * bS1 + z2 * bS2;
    long long coff  = (long long)z1 * cS1  + (long long)z2 * cS2;
    long long ctoff = (long long)z1 * ctS1 + (long long)z2 * ctS2;

    int tid    = threadIdx.x;
    int wave   = tid >> 5;
    int lane   = tid & 31;
    int lane15 = lane & 15;
    int half   = lane >> 4;

    int tileM = blockIdx.y * 64 + wave * 16;
    int tileN = blockIdx.x * 64;
    int mrow  = tileM + lane15;

    // cooperative B-panel stage: thread -> 32B of the 64x32 panel (2 async b128)
    int srow  = tid >> 1;        // 0..63
    int spart = (tid & 1) * 16;  // element offset within row
    int sn    = tileN + srow;
    if (!FULL) sn = (sn < N) ? sn : (N - 1);   // clamp keeps loads in-bounds
    const __bf16* srcrow = Bm + (long long)sn * ldb + spart;

    bool valid[4];
    #pragma unroll
    for (int j = 0; j < 4; ++j) valid[j] = FULL || ((tileN + j * 16) < N);

    v8f acc[4];
    #pragma unroll
    for (int j = 0; j < 4; ++j) acc[j] = (v8f){0.f,0.f,0.f,0.f,0.f,0.f,0.f,0.f};

    // prime buffer 0
    {
        unsigned off = (unsigned)(size_t)&bpanel[0][srow][spart];
        async_copy_b128(srcrow, off);
        async_copy_b128(srcrow + 8, off + 16);
    }
    wait_async0();
    __syncthreads();

    const __bf16* arow = A + (long long)mrow * lda;
    int buf = 0;
    for (int k0 = 0; k0 < K; k0 += 32) {
        // issue async stage of next panel into the other buffer
        if (k0 + 32 < K) {
            const __bf16* g = srcrow + k0 + 32;
            unsigned off = (unsigned)(size_t)&bpanel[buf ^ 1][srow][spart];
            async_copy_b128(g, off);
            async_copy_b128(g + 8, off + 16);
        }

        // ---- A fragment: one contiguous 32B run per lane (k = 16*half + s, matches B) ----
        const __bf16* ap = arow + k0 + half * 16;
        v8bf lo = *(const v8bf*)(ap);
        v8bf hi = *(const v8bf*)(ap + 8);
        __builtin_prefetch((const void*)(ap + 64), 0, 1);
        v16bf a = __builtin_shufflevector(lo, hi, 0,1,2,3,4,5,6,7,8,9,10,11,12,13,14,15);

        // ---- 4 B fragments from LDS + 4 WMMA ----
        #pragma unroll
        for (int j = 0; j < 4; ++j) {
            if (!FULL && !valid[j]) continue;
            v16bf b = *(const v16bf*)&bpanel[buf][j * 16 + lane15][half * 16];
            acc[j] = __builtin_amdgcn_wmma_f32_16x16x32_bf16(
                false, a, false, b, (short)0, acc[j], false, false);
        }

        wait_async0();
        __syncthreads();
        buf ^= 1;
    }

    bool do_tanh = (flags & FLAG_TANH) != 0;
    #pragma unroll
    for (int j = 0; j < 4; ++j) {
        if (!FULL && !valid[j]) continue;
        int n = tileN + j * 16 + lane15;
        float bv = bias ? bias[n] : 0.f;
        #pragma unroll
        for (int r = 0; r < 8; ++r) {
            int m = tileM + r + 8 * half;            // ISA C-layout
            float v = acc[j][r] + bv;
            if (do_tanh) v = tanhf(v);
            if (Cf) Cf[coff  + (long long)m * ldc  + n] = v;
            if (Cb) Cb[coff  + (long long)m * ldc  + n] = (__bf16)v;
            if (Ct) Ct[ctoff + (long long)n * ldct + m] = (__bf16)v;
        }
    }
}

// ---------------- persistent GRU layer (one workgroup, 32 wave32s) ----------------
// LDS: h f32 (64KB) + h bf16 shadow (32KB) + gh f32 (192KB) = 288KB of the WGP's 320KB.
__global__ __launch_bounds__(1024, 1)
void gru_layer(const float* __restrict__ xproj,   // [B*S, 3H] f32
               const __bf16* __restrict__ whh,    // [3H, H] bf16
               const float* __restrict__ bhh,     // [3H]
               __bf16* __restrict__ yout,         // row (b*S+s)*yld
               long long yld)
{
    __shared__ float  h_lds[BB][HID];
    __shared__ __align__(32) __bf16 h_bf[BB][HID];
    __shared__ float  gh_lds[BB][3 * HID];

    int tid    = threadIdx.x;
    int wave   = tid >> 5;
    int lane   = tid & 31;
    int lane15 = lane & 15;
    int half   = lane >> 4;

    for (int i = tid; i < BB * HID; i += 1024) {
        ((float*)h_lds)[i] = 0.f;
        ((__bf16*)h_bf)[i] = (__bf16)0.f;
    }
    __syncthreads();

    for (int s = 0; s < SS; ++s) {
        // ---- gh = h @ whh^T + bhh : tiles 2(M) x 96(N), 6 per wave ----
        for (int t = wave * 6; t < wave * 6 + 6; ++t) {
            int tm = t & 1;
            int tn = t >> 1;
            int m    = tm * 16 + lane15;
            int ncol = tn * 16 + lane15;
            v8f acc = (v8f){0.f,0.f,0.f,0.f,0.f,0.f,0.f,0.f};
            const __bf16* brow = whh + (long long)ncol * HID;
            for (int k0 = 0; k0 < HID; k0 += 32) {
                v8bf lo = *(const v8bf*)(&h_bf[m][k0 + half * 16]);
                v8bf hi = *(const v8bf*)(&h_bf[m][k0 + half * 16 + 8]);
                v16bf a = __builtin_shufflevector(lo, hi,
                              0,1,2,3,4,5,6,7,8,9,10,11,12,13,14,15);
                v16bf b = *(const v16bf*)(brow + k0 + half * 16);
                acc = __builtin_amdgcn_wmma_f32_16x16x32_bf16(
                    false, a, false, b, (short)0, acc, false, false);
            }
            float bv = bhh[ncol];
            #pragma unroll
            for (int r = 0; r < 8; ++r)
                gh_lds[tm * 16 + r + 8 * half][ncol] = acc[r] + bv;
        }
        __syncthreads();

        // ---- gates: r,z,n -> h_new; write y bf16 ----
        for (int i = tid; i < BB * HID; i += 1024) {
            int b = i >> 9, j = i & 511;
            const float* xp = xproj + ((long long)b * SS + s) * (3 * HID);
            float rg = 1.f / (1.f + __expf(-(xp[j] + gh_lds[b][j])));
            float zg = 1.f / (1.f + __expf(-(xp[HID + j] + gh_lds[b][HID + j])));
            float ng = tanhf(xp[2 * HID + j] + rg * gh_lds[b][2 * HID + j]);
            float hn = (1.f - zg) * ng + zg * h_lds[b][j];
            h_lds[b][j] = hn;
            __bf16 hb = (__bf16)hn;
            h_bf[b][j] = hb;
            yout[((long long)b * SS + s) * yld + j] = hb;
        }
        __syncthreads();
    }
}

// ---------------- wave-per-row softmax, f32 in -> bf16 out ----------------
__global__ void softmax_rows(const float* __restrict__ src, __bf16* __restrict__ dst, int ncols)
{
    int row  = blockIdx.x * (blockDim.x >> 5) + (threadIdx.x >> 5);
    int lane = threadIdx.x & 31;
    const float* p = src + (long long)row * ncols;
    float mx = -__builtin_inff();
    for (int c = lane; c < ncols; c += 32) mx = fmaxf(mx, p[c]);
    for (int o = 16; o; o >>= 1) mx = fmaxf(mx, __shfl_xor(mx, o, 32));
    float sum = 0.f;
    for (int c = lane; c < ncols; c += 32) sum += __expf(p[c] - mx);
    for (int o = 16; o; o >>= 1) sum += __shfl_xor(sum, o, 32);
    float inv = 1.f / sum;
    __bf16* q = dst + (long long)row * ncols;
    for (int c = lane; c < ncols; c += 32) q[c] = (__bf16)(__expf(p[c] - mx) * inv);
}

// ---------------- log_softmax + argmax, one block per row ----------------
__global__ __launch_bounds__(256)
void logsoftmax_argmax(const float* __restrict__ logits, float* __restrict__ out, int C)
{
    __shared__ float smax[256];
    __shared__ int   sidx[256];
    __shared__ float ssum[256];
    int row = blockIdx.x;
    int t   = threadIdx.x;
    const float* p = logits + (long long)row * C;

    float mx = -__builtin_inff(); int mi = 0x7fffffff;
    for (int c = t; c < C; c += 256) {
        float v = p[c];
        if (v > mx || (v == mx && c < mi)) { mx = v; mi = c; }
    }
    smax[t] = mx; sidx[t] = mi;
    __syncthreads();
    for (int s2 = 128; s2; s2 >>= 1) {
        if (t < s2) {
            float v = smax[t + s2]; int i2 = sidx[t + s2];
            if (v > smax[t] || (v == smax[t] && i2 < sidx[t])) { smax[t] = v; sidx[t] = i2; }
        }
        __syncthreads();
    }
    float rmax = smax[0];
    float sum = 0.f;
    for (int c = t; c < C; c += 256) sum += __expf(p[c] - rmax);
    ssum[t] = sum;
    __syncthreads();
    for (int s2 = 128; s2; s2 >>= 1) { if (t < s2) ssum[t] += ssum[t + s2]; __syncthreads(); }
    float lse = rmax + __logf(ssum[0]);

    float* lg = out + BB * SS + (long long)row * C;
    for (int c = t; c < C; c += 256) lg[c] = p[c] - lse;
    if (t == 0) out[row] = (float)sidx[0];
}

// ---------------- host orchestration ----------------
extern "C" void kernel_launch(void* const* d_in, const int* in_sizes, int n_in,
                              void* d_out, int out_size, void* d_ws, size_t ws_size,
                              hipStream_t stream)
{
    (void)in_sizes; (void)n_in; (void)out_size; (void)ws_size;
    const int*   tokens  = (const int*)  d_in[0];
    const float* lst     = (const float*)d_in[1];
    const float* emb     = (const float*)d_in[2];
    const float* w_ih    = (const float*)d_in[3];
    const float* w_hh    = (const float*)d_in[4];
    const float* b_ih    = (const float*)d_in[5];
    const float* b_hh    = (const float*)d_in[6];
    const float* wq      = (const float*)d_in[7];
    const float* bq      = (const float*)d_in[8];
    const float* wv      = (const float*)d_in[9];
    const float* bv      = (const float*)d_in[10];
    const float* w_attn  = (const float*)d_in[11];
    const float* b_attn  = (const float*)d_in[12];
    const float* fc_w    = (const float*)d_in[13];
    const float* fc_b    = (const float*)d_in[14];
    float* out = (float*)d_out;

    const long long MBS = (long long)BB * SS;      // 4096
    const long long MBT = (long long)BB * TT;      // 32768
    const long long H3  = 3 * HID;                 // 1536
    const long long QKD = NUM_HEAD * ATTN_DIM;     // 512

    // ---- workspace carve-out ----
    char* p = (char*)d_ws;
    auto alloc = [&](size_t bytes) -> char* {
        char* r = p; p += (bytes + 255) & ~(size_t)255; return r;
    };
    __bf16* wih_bf   = (__bf16*)alloc(NUM_LAYERS * H3 * HID * 2);
    __bf16* whh_bf   = (__bf16*)alloc(NUM_LAYERS * H3 * HID * 2);
    __bf16* wq_bf    = (__bf16*)alloc(QKD * HID * 2);
    __bf16* wv_bf    = (__bf16*)alloc(QKD * HID * 2);
    __bf16* wat_bf   = (__bf16*)alloc(HID * (HID + QKD) * 2);
    __bf16* fcw_bf   = (__bf16*)alloc((size_t)NUM_CLASS * HID * 2);
    __bf16* x0_bf    = (__bf16*)alloc(MBS * HID * 2);
    __bf16* lst_bf   = (__bf16*)alloc(MBT * HID * 2);
    __bf16* y0_bf    = (__bf16*)alloc(MBS * HID * 2);
    __bf16* comb_bf  = (__bf16*)alloc(MBS * (HID + QKD) * 2);
    __bf16* q_bf     = (__bf16*)alloc(MBS * QKD * 2);
    __bf16* v_bf     = (__bf16*)alloc(MBT * QKD * 2);
    __bf16* vt_bf    = (__bf16*)alloc(MBT * QKD * 2);            // V transposed [b, d, t]
    float*  xproj    = (float*) alloc(MBS * H3 * 4);             // reused by attnout_bf
    float*  score    = (float*) alloc((size_t)BB * NUM_HEAD * SS * TT * 4); // reused by logits
    __bf16* align_bf = (__bf16*)alloc((size_t)BB * NUM_HEAD * SS * TT * 2);
    __bf16* atn_bf   = (__bf16*)xproj;                            // overlay (xproj dead)
    float*  logits   = (float*) score;                            // overlay (score dead)

    auto cast = [&](const float* s, __bf16* d, long long n) {
        cast_f32_bf16<<<dim3((unsigned)((n + 255) / 256)), dim3(256), 0, stream>>>(s, d, n);
    };
    auto gemm = [&](const __bf16* A, long long lda, long long aS1, long long aS2,
                    const __bf16* Bm, long long ldb, long long bS1, long long bS2,
                    const float* bias, float* Cf, __bf16* Cb, __bf16* Ct,
                    long long ldc, long long cS1, long long cS2,
                    long long ldct, long long ctS1, long long ctS2,
                    int M, int N, int K, int nbatch, int batch2, int flags) {
        dim3 grid((unsigned)((N + 63) / 64), (unsigned)(M / 64), (unsigned)nbatch);
        if ((N & 63) == 0)
            gemm_wmma<true><<<grid, dim3(128), 0, stream>>>(
                A, lda, aS1, aS2, Bm, ldb, bS1, bS2, bias, Cf, Cb, Ct,
                ldc, cS1, cS2, ldct, ctS1, ctS2, M, N, K, batch2, flags);
        else
            gemm_wmma<false><<<grid, dim3(128), 0, stream>>>(
                A, lda, aS1, aS2, Bm, ldb, bS1, bS2, bias, Cf, Cb, Ct,
                ldc, cS1, cS2, ldct, ctS1, ctS2, M, N, K, batch2, flags);
    };

    // 1) bf16 weight / activation copies
    cast(w_ih,   wih_bf, NUM_LAYERS * H3 * HID);
    cast(w_hh,   whh_bf, NUM_LAYERS * H3 * HID);
    cast(wq,     wq_bf,  QKD * HID);
    cast(wv,     wv_bf,  QKD * HID);
    cast(w_attn, wat_bf, HID * (HID + QKD));
    cast(fc_w,   fcw_bf, (long long)NUM_CLASS * HID);
    cast(lst,    lst_bf, MBT * HID);

    // 2) embedding gather -> bf16
    gather_emb<<<dim3((unsigned)((MBS * HID + 255) / 256)), dim3(256), 0, stream>>>(
        tokens, emb, x0_bf, (int)(MBS * HID));

    // 3) layer 0: x_proj = x0 @ w_ih0^T + b_ih0 ; persistent GRU -> y0 (bf16)
    gemm(x0_bf, HID, 0, 0, wih_bf, HID, 0, 0, b_ih, xproj, nullptr, nullptr,
         H3, 0, 0, 0, 0, 0, (int)MBS, (int)H3, HID, 1, 1, 0);
    gru_layer<<<dim3(1), dim3(1024), 0, stream>>>(xproj, whh_bf, b_hh, y0_bf, HID);

    // 4) layer 1: x_proj = y0 @ w_ih1^T + b_ih1 ; GRU writes into combined[:,512:1024]
    gemm(y0_bf, HID, 0, 0, wih_bf + H3 * HID, HID, 0, 0, b_ih + H3, xproj, nullptr, nullptr,
         H3, 0, 0, 0, 0, 0, (int)MBS, (int)H3, HID, 1, 1, 0);
    gru_layer<<<dim3(1), dim3(1024), 0, stream>>>(
        xproj, whh_bf + H3 * HID, b_hh + H3, comb_bf + HID, HID + QKD);

    // 5) Q = out @ wq^T + bq (bf16)
    gemm(comb_bf + HID, HID + QKD, 0, 0, wq_bf, HID, 0, 0, bq, nullptr, q_bf, nullptr,
         QKD, 0, 0, 0, 0, 0, (int)MBS, (int)QKD, HID, 1, 1, 0);

    // 6) V = listener @ wv^T + bv, batched over b (M=T per batch) so the epilogue
    //    also emits V^T [b, d, t] for the context GEMM (keeps everything NT).
    gemm(lst_bf, HID, (long long)TT * HID, 0,
         wv_bf, HID, 0, 0, bv,
         nullptr, v_bf, vt_bf,
         QKD, (long long)TT * QKD, 0,
         TT, QKD * TT, 0,
         TT, (int)QKD, HID, BB, 1, 0);

    // 7) score[b,n,s,t] = q . v  (batched over z=b*NH+n; NT, v rows contiguous in d)
    gemm(q_bf, QKD, (long long)SS * QKD, ATTN_DIM,
         v_bf, QKD, (long long)TT * QKD, ATTN_DIM,
         nullptr, score, nullptr, nullptr,
         TT, (long long)NUM_HEAD * SS * TT, (long long)SS * TT,
         0, 0, 0,
         SS, TT, ATTN_DIM, BB * NUM_HEAD, NUM_HEAD, 0);

    // 8) softmax over t -> bf16 align
    softmax_rows<<<dim3((unsigned)(BB * NUM_HEAD * SS / 8)), dim3(256), 0, stream>>>(
        score, align_bf, TT);

    // 9) ctx = align @ (V^T)^T : NT GEMM against vt rows (d, contiguous t),
    //    written into combined[:,0:512]
    gemm(align_bf, TT, (long long)NUM_HEAD * SS * TT, (long long)SS * TT,
         vt_bf, TT, QKD * TT, (long long)ATTN_DIM * TT,
         nullptr, nullptr, comb_bf, nullptr,
         HID + QKD, (long long)SS * (HID + QKD), ATTN_DIM,
         0, 0, 0,
         SS, ATTN_DIM, TT, BB * NUM_HEAD, NUM_HEAD, 0);

    // 10) attn_out = tanh(combined @ w_attn^T + b_attn) -> bf16
    gemm(comb_bf, HID + QKD, 0, 0, wat_bf, HID + QKD, 0, 0, b_attn,
         nullptr, atn_bf, nullptr,
         HID, 0, 0, 0, 0, 0, (int)MBS, HID, HID + QKD, 1, 1, FLAG_TANH);

    // 11) logits = attn_out @ fc_w^T + fc_b  (f32)
    gemm(atn_bf, HID, 0, 0, fcw_bf, HID, 0, 0, fc_b, logits, nullptr, nullptr,
         NUM_CLASS, 0, 0, 0, 0, 0, (int)MBS, NUM_CLASS, HID, 1, 1, 0);

    // 12) log_softmax + argmax -> d_out (hyp[4096] then logit[4096*2000])
    logsoftmax_argmax<<<dim3((unsigned)MBS), dim3(256), 0, stream>>>(logits, out, NUM_CLASS);
}
